// ConvShiftLayer_50878182588449
// MI455X (gfx1250) — matrix operand order repaired and checked by reading, not compiled, not run on hardware
//
#include <hip/hip_runtime.h>
#include <hip/hip_bf16.h>
#include <math.h>

typedef __attribute__((ext_vector_type(8)))  __bf16 bf16x8;
typedef __attribute__((ext_vector_type(16))) __bf16 bf16x16;
typedef __attribute__((ext_vector_type(8)))  float  f32x8;

#define IN_DIM 768
#define FD     1024
#define SEQ    1024
#define NBATCH 4
#define M_TOT  (NBATCH * SEQ)      // 4096 rows of x / a / z

// GEMM tiling: block 128(M) x 128(N), K-step 32; 8 wave32s, wave tile 32x64.
#define BM 128
#define BN 128
#define BK 32
#define PITCH16 40                 // bf16 per LDS row (80 B: 16B-aligned, conflict-free)
#define PITCH32 20

// Conv tiling
#define L_ROWS 128

__device__ __forceinline__ unsigned int f2bf(float f) {
    unsigned int u = __float_as_uint(f);
    u += 0x7FFFu + ((u >> 16) & 1u);          // round-to-nearest-even
    return u >> 16;
}
__device__ __forceinline__ unsigned int pack2bf(float lo, float hi) {
    return (f2bf(lo) & 0xFFFFu) | (f2bf(hi) << 16);
}

// Branchless tanh: one v_exp_f32 + one v_rcp_f32, no exec-mask branches.
__device__ __forceinline__ float fast_tanh(float x) {
    const float ax = __builtin_fabsf(x);
    const float e  = __builtin_amdgcn_exp2f(ax * -2.885390081777927f); // exp(-2|x|)
    const float r  = (1.0f - e) * __builtin_amdgcn_rcpf(1.0f + e);
    return __builtin_copysignf(r, x);
}

// ---------------------------------------------------------------------------
// Kernel 1: a = tanh(x @ W + b), z = (pre > 0), bf16 WMMA / f32 accum.
// Double-buffered LDS, one barrier per K-step; base-pointer bumping with
// constant 24-bit instruction offsets for all staging loads and output stores.
// ---------------------------------------------------------------------------
__global__ __launch_bounds__(256) void gemm_tanh_step_kernel(
    const float* __restrict__ x,   // [4096, 768]
    const float* __restrict__ W,   // [768, 1024]
    const float* __restrict__ bias,// [1024]
    float* __restrict__ a_out,     // [4096, 1024]
    float* __restrict__ z_out)     // [4096, 1024]
{
    __shared__ __align__(16) unsigned int As[2][BM * PITCH32]; // 2 x 10240 B
    __shared__ __align__(16) unsigned int Bs[2][BN * PITCH32]; // 2 x 10240 B

    const int tid   = threadIdx.x;
    const int lane  = tid & 31;
    const int r     = lane & 15;
    const int h     = lane >> 4;
    const int wid   = tid >> 5;
    const int waveM = wid & 3;     // 0..3 -> 32-row strips
    const int waveN = wid >> 2;    // 0..1 -> 64-col strips

    const int mBase = blockIdx.y * BM;
    const int nBase = blockIdx.x * BN;

    // ---- staging base pointers (bumped each K-step; all else is immediate)
    // A: thread covers rows {tid>>4 + 16*i}, K-pair kp = tid&15.
    const char* aPtr = (const char*)(x + (size_t)(mBase + (tid >> 4)) * IN_DIM + (tid & 15) * 2);
    // B: thread covers cols n = tid&127, K rows {2*(tid>>7) + 4*i} (+1 for hi).
    const char* bPtr = (const char*)(W + (size_t)(tid >> 7) * 2 * FD + nBase + (tid & 127));

    // ---- invariant LDS addresses
    unsigned int* aSt0 = &As[0][(tid >> 4) * PITCH32 + (tid & 15)];
    unsigned int* bSt0 = &Bs[0][(tid & 127) * PITCH32 + (tid >> 7)];
    const int bufStride = BM * PITCH32;       // u32 elements between buffers

    f32x8 acc[2][4] = {};
    unsigned int aReg[8], bReg[8];

    auto loadRegs = [&]() {
        #pragma unroll
        for (int i = 0; i < 8; ++i) {          // A: float2, +16 rows per i
            const float2 f2 = *(const float2*)(aPtr + (size_t)i * (16 * IN_DIM * 4));
            aReg[i] = pack2bf(f2.x, f2.y);
        }
        #pragma unroll
        for (int i = 0; i < 8; ++i) {          // B: two f32, +4 K rows per i
            float lo = *(const float*)(bPtr + (size_t)i * (4 * FD * 4));
            float hi = *(const float*)(bPtr + (size_t)i * (4 * FD * 4) + FD * 4);
            bReg[i] = pack2bf(lo, hi);
        }
        aPtr += BK * 4;                        // +128 B
        bPtr += (size_t)BK * FD * 4;           // +128 KB
    };
    auto storeLds = [&](int p) {
        #pragma unroll
        for (int i = 0; i < 8; ++i)
            aSt0[p * bufStride + i * (16 * PITCH32)] = aReg[i];
        #pragma unroll
        for (int i = 0; i < 8; ++i)
            bSt0[p * bufStride + i * (32 * PITCH32) + 2 * (i & 0)] = bReg[i]; // kp += 2 per i
    };
    // NOTE on Bs store offsets: n fixed = tid&127, kp = (tid>>7) + 2*i
    //   -> element offset = 2*i (u32) within the row; rewrite properly below.
    auto storeLdsB = [&](int p) {
        #pragma unroll
        for (int i = 0; i < 8; ++i)
            bSt0[p * bufStride + 2 * i] = bReg[i];
    };

    auto compute = [&](int p) {
        bf16x16 afr[2], bfr[4];
        const unsigned short* As16 = (const unsigned short*)As[p];
        const unsigned short* Bs16 = (const unsigned short*)Bs[p];
        #pragma unroll
        for (int mt = 0; mt < 2; ++mt) {
            // A lane r: row M = r; K runs {8h..8h+7} and {16+8h..23+8h}
            const unsigned short* pa = As16 + (waveM * 32 + mt * 16 + r) * PITCH16;
            bf16x8 lo = *(const bf16x8*)(pa + 8 * h);
            bf16x8 hi = *(const bf16x8*)(pa + 16 + 8 * h);
            afr[mt] = __builtin_shufflevector(lo, hi, 0,1,2,3,4,5,6,7,8,9,10,11,12,13,14,15);
        }
        #pragma unroll
        for (int nt = 0; nt < 4; ++nt) {
            // B lane r: column N = r; K runs {8h..8h+7} and {16+8h..23+8h}
            const unsigned short* pb = Bs16 + (waveN * 64 + nt * 16 + r) * PITCH16;
            bf16x8 lo = *(const bf16x8*)(pb + 8 * h);
            bf16x8 hi = *(const bf16x8*)(pb + 16 + 8 * h);
            bfr[nt] = __builtin_shufflevector(lo, hi, 0,1,2,3,4,5,6,7,8,9,10,11,12,13,14,15);
        }
        #pragma unroll
        for (int mt = 0; mt < 2; ++mt)
            #pragma unroll
            for (int nt = 0; nt < 4; ++nt)
                acc[mt][nt] = __builtin_amdgcn_wmma_f32_16x16x32_bf16(
                    false, afr[mt], false, bfr[nt], (short)0, acc[mt][nt], false, false);
    };

    // ---- pipelined main loop ----------------------------------------------
    loadRegs();
    storeLds(0);
    storeLdsB(0);
    __syncthreads();
    int p = 0;
    #pragma unroll 1
    for (int k0 = BK; k0 < IN_DIM; k0 += BK) {
        loadRegs();          // global prefetch flies under the WMMAs below
        compute(p);          // read buffer p
        storeLds(p ^ 1);     // waits loadcnt, fills the other buffer
        storeLdsB(p ^ 1);
        __syncthreads();
        p ^= 1;
    }
    compute(p);

    // ---- epilogue: base pointers + immediate offsets -----------------------
    // C layout: lane (r,h), VGPR e -> (M = e + 8h, N = r)
    const size_t outBase = (size_t)(mBase + waveM * 32 + 8 * h) * FD
                         + nBase + waveN * 64 + r;
    float* aB = a_out + outBase;
    float* zB = z_out + outBase;
    const float* bB = bias + nBase + waveN * 64 + r;
    #pragma unroll
    for (int nt = 0; nt < 4; ++nt) {
        const float bv = bB[nt * 16];
        #pragma unroll
        for (int mt = 0; mt < 2; ++mt) {
            #pragma unroll
            for (int e = 0; e < 8; ++e) {
                const float pre = acc[mt][nt][e] + bv;
                const float av  = fast_tanh(pre);
                const float zv  = pre > 0.0f ? 1.0f : 0.0f;
                const int off = nt * 16 + (mt * 16 + e) * FD;   // all-immediate
                aB[off] = av;
                zB[off] = zv;
            }
        }
    }
}

// ---------------------------------------------------------------------------
// Kernel 2: z_conv[s,f] = sum_{d=0..7} z[s+4-d, (f+d) mod 1024]  (rows OOB -> 0)
// Skewed-column 8-tap box filter: read channel at row t == write channel at
// emission s = t-4, so each element is 1 coalesced load + 1 coalesced store.
// ---------------------------------------------------------------------------
__global__ __launch_bounds__(256) void shift_conv_kernel(
    const float* __restrict__ z,   // [NBATCH, 1024, 1024]
    float* __restrict__ zc)        // [NBATCH, 1024, 1024]
{
    const int c  = blockIdx.x * 256 + threadIdx.x;    // skewed column, 0..1023
    const int r0 = blockIdx.y * L_ROWS;
    const float* zb  = z  + (size_t)blockIdx.z * SEQ * FD;
    float*       zcb = zc + (size_t)blockIdx.z * SEQ * FD;

    float buf[8];
    #pragma unroll
    for (int i = 0; i < 8; ++i) buf[i] = 0.0f;
    float sum = 0.0f;

    for (int tt = 0; tt < L_ROWS + 8; tt += 8) {
        #pragma unroll
        for (int u = 0; u < 8; ++u) {             // static ring index = u
            const int t  = r0 - 3 + tt + u;
            const int ch = (c - t) & (FD - 1);
            float v = 0.0f;
            if (t >= 0 && t < SEQ && t <= r0 + L_ROWS + 3)
                v = zb[(size_t)t * FD + ch];
            sum -= buf[u];
            buf[u] = v;
            sum += v;
            const int s = t - 4;
            if (s >= r0 && s < r0 + L_ROWS)
                zcb[(size_t)s * FD + ch] = sum;
        }
    }
}

// ---------------------------------------------------------------------------
extern "C" void kernel_launch(void* const* d_in, const int* in_sizes, int n_in,
                              void* d_out, int out_size, void* d_ws, size_t ws_size,
                              hipStream_t stream) {
    const float* x = (const float*)d_in[0];   // (4,1024,768)
    const float* W = (const float*)d_in[1];   // (768,1024)
    const float* b = (const float*)d_in[2];   // (1024,)

    float* out    = (float*)d_out;
    const size_t PLANE = (size_t)NBATCH * SEQ * FD;   // 4,194,304
    float* a_out  = out;
    float* z_out  = out + PLANE;
    float* zc_out = out + 2 * PLANE;

    dim3 g1(FD / BN, M_TOT / BM);             // (8, 32)
    gemm_tanh_step_kernel<<<g1, 256, 0, stream>>>(x, W, b, a_out, z_out);

    dim3 g2(FD / 256, SEQ / L_ROWS, NBATCH);  // (4, 8, 4)
    shift_conv_kernel<<<g2, 256, 0, stream>>>(z_out, zc_out);
}
